// MyModel_32839319945756
// MI455X (gfx1250) — matrix operand order repaired and checked by reading, not compiled, not run on hardware
//
#include <hip/hip_runtime.h>

// ---------------- problem constants ----------------
#define BSZ    256
#define TSEQ   1024
#define FIN    73
#define KPAD0  96          // FIN padded to multiple of 32
#define HID    256
#define NLAY   3
#define OUTDIM 35
#define NEG_SLOPE 0.01f

#define NBLK   24          // 3 layers * 8 hidden-blocks
#define NTH    256         // 8 wave32s

// ---------------- WMMA types ----------------
typedef __attribute__((ext_vector_type(16))) __bf16 v16bf;
typedef __attribute__((ext_vector_type(8)))  float  v8f;

__device__ __forceinline__ v8f wmma_bf16(v16bf a, v16bf b, v8f c) {
    return __builtin_amdgcn_wmma_f32_16x16x32_bf16(false, a, false, b, (short)0, c, false, false);
}

__device__ __forceinline__ __bf16 f2bf(float f) {
    unsigned u = __builtin_bit_cast(unsigned, f);
    u += 0x7FFFu + ((u >> 16) & 1u);               // round-to-nearest-even
    unsigned short h = (unsigned short)(u >> 16);
    return __builtin_bit_cast(__bf16, h);
}

__device__ __forceinline__ float sigm_(float x) { return 1.f / (1.f + __expf(-x)); }
__device__ __forceinline__ float tanh_(float x) {
    float e = __expf(2.f * x);
    return (e - 1.f) / (e + 1.f);
}

// ---------------- workspace layout (byte offsets) ----------------
#define SZ_TILE   (512 * 2)                     // one 16x32 / 32x16 bf16 fragment tile
#define WS_WIH0   256
#define SZ_WIH0   (64 * 3 * SZ_TILE)            // 64 n-tiles x 3 k-tiles (Kpad=96)
#define WS_WHH0   (WS_WIH0 + SZ_WIH0)
#define SZ_W8     (64 * 8 * SZ_TILE)            // 64 n-tiles x 8 k-tiles (K=256)
#define WS_WIH1   (WS_WHH0 + SZ_W8)
#define WS_WHH1   (WS_WIH1 + SZ_W8)
#define WS_WIH2   (WS_WHH1 + SZ_W8)
#define WS_WHH2   (WS_WIH2 + SZ_W8)
#define WS_FCW    (WS_WHH2 + SZ_W8)
#define SZ_FCW    (3 * 8 * SZ_TILE)             // 48 cols (O=35 padded) x K=256
#define WS_HFRAG  (WS_FCW + SZ_FCW)
#define SZ_HF     (16 * 8 * SZ_TILE)            // 16 M-tiles x 8 K-tiles, bf16
#define WS_HLRELU (WS_HFRAG + 6 * SZ_HF)        // leaky-relu(h2[T-1]) in A-frag order

#define HOFF      (BSZ * OUTDIM)                // d_out offsets (floats)
#define COFF      (HOFF + NLAY * BSZ * HID)

__device__ __forceinline__ __bf16* hbuf(char* ws, int l, int par) {
    return (__bf16*)(ws + WS_HFRAG + (size_t)(l * 2 + par) * SZ_HF);
}

// ---------------- grid barrier (persistent kernel) ----------------
__device__ void grid_barrier(unsigned* cnt, unsigned* gen, unsigned* lgen) {
    __syncthreads();
    if (threadIdx.x == 0) {
        unsigned target = ++(*lgen);
        __threadfence();
        unsigned a = atomicAdd(cnt, 1u);
        if (a == NBLK - 1u) {
            atomicExch(cnt, 0u);
            __threadfence();
            atomicAdd(gen, 1u);
        } else {
            while (__atomic_load_n(gen, __ATOMIC_RELAXED) < target)
                __builtin_amdgcn_s_sleep(1);
            __threadfence();
        }
    }
    __syncthreads();
}

// fp32 W[N,K] (row-major, leading dim ld) -> bf16 B-operand fragments.
// B tile (32K x 16N): lane = ((k>>3)&1)*16 + n ; element = (k>>4)*8 + (k&7).
__device__ void swizzleB(const float* W, int Nsrc, int Ksrc, int ld,
                         __bf16* dst, int NT, int KT, int gtid) {
    int total = NT * KT * 512;
    for (int e = gtid; e < total; e += NBLK * NTH) {
        int tile = e >> 9, off = e & 511;
        int nt = tile / KT, kt = tile - nt * KT;
        int ln = off >> 4, pos = off & 15;
        int v = pos >> 1, bb = pos & 1;
        int k = kt * 32 + ((v >> 2) << 4) + ((ln >> 4) << 3) + ((v & 3) << 1) + bb;
        int n = nt * 16 + (ln & 15);
        float val = (k < Ksrc && n < Nsrc) ? W[(size_t)n * ld + k] : 0.f;
        dst[e] = f2bf(val);
    }
}

__global__ __launch_bounds__(NTH) void lstm3_wmma_kernel(
    const float* __restrict__ x,
    const float* __restrict__ Wih0, const float* __restrict__ Whh0,
    const float* __restrict__ bih0, const float* __restrict__ bhh0,
    const float* __restrict__ Wih1, const float* __restrict__ Whh1,
    const float* __restrict__ bih1, const float* __restrict__ bhh1,
    const float* __restrict__ Wih2, const float* __restrict__ Whh2,
    const float* __restrict__ bih2, const float* __restrict__ bhh2,
    const float* __restrict__ fcw,  const float* __restrict__ fcb,
    float* __restrict__ dout, char* __restrict__ ws)
{
    __shared__ __bf16 sA[16 * 3 * 512];          // x_t A-fragments (layer 0), 48 KB

    const int gid   = blockIdx.x;
    const int layer = gid >> 3;
    const int jb    = gid & 7;                   // owns hidden units [32*jb, 32*jb+32)
    const int wave  = threadIdx.x >> 5;          // owns batch rows [32*wave, 32*wave+32)
    const int lane  = threadIdx.x & 31;
    const int gtid  = gid * NTH + threadIdx.x;

    unsigned* cnt = (unsigned*)ws;
    unsigned* gen = cnt + 1;
    unsigned  lgen = 0;

    // ---------------- phase 0: weight swizzle + h-buffer zero ----------------
    swizzleB(Wih0, 1024, FIN, FIN, (__bf16*)(ws + WS_WIH0), 64, 3, gtid);
    swizzleB(Whh0, 1024, HID, HID, (__bf16*)(ws + WS_WHH0), 64, 8, gtid);
    swizzleB(Wih1, 1024, HID, HID, (__bf16*)(ws + WS_WIH1), 64, 8, gtid);
    swizzleB(Whh1, 1024, HID, HID, (__bf16*)(ws + WS_WHH1), 64, 8, gtid);
    swizzleB(Wih2, 1024, HID, HID, (__bf16*)(ws + WS_WIH2), 64, 8, gtid);
    swizzleB(Whh2, 1024, HID, HID, (__bf16*)(ws + WS_WHH2), 64, 8, gtid);
    swizzleB(fcw,  OUTDIM, HID, HID, (__bf16*)(ws + WS_FCW), 3, 8, gtid);
    {
        unsigned* zp = (unsigned*)(ws + WS_HFRAG);
        for (unsigned e = gtid; e < (7u * SZ_HF) / 4u; e += NBLK * NTH) zp[e] = 0u;
    }
    grid_barrier(cnt, gen, &lgen);

    // ---------------- per-block constants ----------------
    const float* bihL = (layer == 0) ? bih0 : (layer == 1) ? bih1 : bih2;
    const float* bhhL = (layer == 0) ? bhh0 : (layer == 1) ? bhh1 : bhh2;
    const __bf16* wihF = (const __bf16*)(ws + ((layer == 0) ? WS_WIH0 : (layer == 1) ? WS_WIH1 : WS_WIH2));
    const __bf16* whhF = (const __bf16*)(ws + ((layer == 0) ? WS_WHH0 : (layer == 1) ? WS_WHH1 : WS_WHH2));
    const int ktIh = (layer == 0) ? 3 : 8;

    const __bf16* wq_ih[8];
    const __bf16* wq_hh[8];
    float bias[8];
    #pragma unroll
    for (int q = 0; q < 8; ++q) {
        int gi = q >> 1, sub = q & 1;
        int ntG = gi * 16 + 2 * jb + sub;                 // z-column tile
        wq_ih[q] = wihF + (size_t)ntG * ktIh * 512;
        wq_hh[q] = whhF + (size_t)ntG * 8 * 512;
        int col = gi * 256 + 32 * jb + 16 * sub + (lane & 15);
        bias[q] = bihL[col] + bhhL[col];
    }

    // persistent cell state: c for (rows 32*wave.., units 32*jb..), D-tile layout
    float creg[2][2][8] = {};

    // ---------------- pipelined recurrence ----------------
    for (int s = 0; s < TSEQ + NLAY - 1; ++s) {
        const int t = s - layer;
        if (t >= 0 && t < TSEQ) {
            v8f acc[2][8] = {};

            // ---- GEMM1: z += x_in @ Wih^T ----
            if (layer == 0) {
                __syncthreads();
                for (int e = threadIdx.x; e < 16 * 3 * 512; e += NTH) {
                    int tile = e >> 9, off = e & 511;
                    int tm = tile / 3, kt = tile - tm * 3;
                    int ln = off >> 4, pos = off & 15;
                    int vp = pos >> 1, bb = pos & 1;
                    int kl = ((vp >> 2) << 4) + ((ln >> 4) << 3) + ((vp & 3) << 1) + bb;
                    int k = kt * 32 + kl;
                    int m = tm * 16 + (ln & 15);
                    float val = (k < FIN) ? x[((size_t)m * TSEQ + t) * FIN + k] : 0.f;
                    sA[e] = f2bf(val);
                }
                __syncthreads();
                #pragma unroll
                for (int kt = 0; kt < 3; ++kt) {
                    v16bf a0 = *(const v16bf*)&sA[((2 * wave)     * 3 + kt) * 512 + lane * 16];
                    v16bf a1 = *(const v16bf*)&sA[((2 * wave + 1) * 3 + kt) * 512 + lane * 16];
                    #pragma unroll
                    for (int q = 0; q < 8; ++q) {
                        v16bf bw = *(const v16bf*)(wq_ih[q] + kt * 512 + lane * 16);
                        acc[0][q] = wmma_bf16(a0, bw, acc[0][q]);
                        acc[1][q] = wmma_bf16(a1, bw, acc[1][q]);
                    }
                }
            } else {
                const __bf16* aIn = hbuf(ws, layer - 1, (s - 1) & 1);
                for (int kt = 0; kt < 8; ++kt) {
                    v16bf a0 = *(const v16bf*)(aIn + (size_t)((2 * wave)     * 8 + kt) * 512 + lane * 16);
                    v16bf a1 = *(const v16bf*)(aIn + (size_t)((2 * wave + 1) * 8 + kt) * 512 + lane * 16);
                    #pragma unroll
                    for (int q = 0; q < 8; ++q) {
                        v16bf bw = *(const v16bf*)(wq_ih[q] + kt * 512 + lane * 16);
                        acc[0][q] = wmma_bf16(a0, bw, acc[0][q]);
                        acc[1][q] = wmma_bf16(a1, bw, acc[1][q]);
                    }
                }
            }

            // ---- GEMM2: z += h_prev @ Whh^T ----
            {
                const __bf16* aH = hbuf(ws, layer, (s - 1) & 1);
                for (int kt = 0; kt < 8; ++kt) {
                    if (kt + 1 < 8)
                        __builtin_prefetch(aH + (size_t)((2 * wave) * 8 + kt + 1) * 512 + lane * 16, 0, 3);
                    v16bf a0 = *(const v16bf*)(aH + (size_t)((2 * wave)     * 8 + kt) * 512 + lane * 16);
                    v16bf a1 = *(const v16bf*)(aH + (size_t)((2 * wave + 1) * 8 + kt) * 512 + lane * 16);
                    #pragma unroll
                    for (int q = 0; q < 8; ++q) {
                        v16bf bw = *(const v16bf*)(wq_hh[q] + kt * 512 + lane * 16);
                        acc[0][q] = wmma_bf16(a0, bw, acc[0][q]);
                        acc[1][q] = wmma_bf16(a1, bw, acc[1][q]);
                    }
                }
            }

            // ---- LSTM cell update (wave-local) + scatter h in A-frag order ----
            __bf16* hOut = hbuf(ws, layer, s & 1);
            __bf16* hLr  = (__bf16*)(ws + WS_HLRELU);
            #pragma unroll
            for (int mt = 0; mt < 2; ++mt) {
                #pragma unroll
                for (int sub = 0; sub < 2; ++sub) {
                    #pragma unroll
                    for (int v = 0; v < 8; ++v) {
                        float zi = acc[mt][0 + sub][v] + bias[0 + sub];
                        float zf = acc[mt][2 + sub][v] + bias[2 + sub];
                        float zg = acc[mt][4 + sub][v] + bias[4 + sub];
                        float zo = acc[mt][6 + sub][v] + bias[6 + sub];
                        float c = sigm_(zf) * creg[mt][sub][v] + sigm_(zi) * tanh_(zg);
                        creg[mt][sub][v] = c;
                        float h = sigm_(zo) * tanh_(c);

                        int mprime = ((lane >> 4) << 3) + v;             // m & 15
                        int lanep  = (((lane >> 3) & 1) << 4) + mprime;  // dest lane
                        int vp     = (sub << 2) + ((lane & 7) >> 1);     // dest packed pair
                        int bp     = lane & 1;
                        int tileM  = 2 * wave + mt;
                        size_t off = ((size_t)(tileM * 8 + jb)) * 512 + lanep * 16 + vp * 2 + bp;
                        hOut[off] = f2bf(h);
                        if (layer == 2 && t == TSEQ - 1)
                            hLr[off] = f2bf(h > 0.f ? h : NEG_SLOPE * h);
                        if (t == TSEQ - 1) {
                            int m = 32 * wave + 16 * mt + ((lane >> 4) << 3) + v;
                            int j = 32 * jb + 16 * sub + (lane & 15);
                            dout[HOFF + ((size_t)layer * BSZ + m) * HID + j] = h;
                            dout[COFF + ((size_t)layer * BSZ + m) * HID + j] = c;
                        }
                    }
                }
            }
        }
        grid_barrier(cnt, gen, &lgen);
    }

    // ---------------- final FC: out = leaky_relu(h2[T-1]) @ fc_w^T + fc_b ----------------
    if (gid == 0) {
        const __bf16* aL = (const __bf16*)(ws + WS_HLRELU);
        const __bf16* fw = (const __bf16*)(ws + WS_FCW);
        v8f facc[2][3] = {};
        for (int kt = 0; kt < 8; ++kt) {
            v16bf a0 = *(const v16bf*)(aL + (size_t)((2 * wave)     * 8 + kt) * 512 + lane * 16);
            v16bf a1 = *(const v16bf*)(aL + (size_t)((2 * wave + 1) * 8 + kt) * 512 + lane * 16);
            #pragma unroll
            for (int nt = 0; nt < 3; ++nt) {
                v16bf bw = *(const v16bf*)(fw + (size_t)(nt * 8 + kt) * 512 + lane * 16);
                facc[0][nt] = wmma_bf16(a0, bw, facc[0][nt]);
                facc[1][nt] = wmma_bf16(a1, bw, facc[1][nt]);
            }
        }
        #pragma unroll
        for (int mt = 0; mt < 2; ++mt) {
            #pragma unroll
            for (int nt = 0; nt < 3; ++nt) {
                int col = nt * 16 + (lane & 15);
                #pragma unroll
                for (int v = 0; v < 8; ++v) {
                    if (col < OUTDIM) {
                        int m = 32 * wave + 16 * mt + ((lane >> 4) << 3) + v;
                        dout[(size_t)m * OUTDIM + col] = facc[mt][nt][v] + fcb[col];
                    }
                }
            }
        }
    }
}

extern "C" void kernel_launch(void* const* d_in, const int* in_sizes, int n_in,
                              void* d_out, int out_size, void* d_ws, size_t ws_size,
                              hipStream_t stream) {
    (void)in_sizes; (void)n_in; (void)out_size; (void)ws_size;
    // reset grid-barrier counters (capture-legal async memset)
    hipMemsetAsync(d_ws, 0, 256, stream);
    lstm3_wmma_kernel<<<NBLK, NTH, 0, stream>>>(
        (const float*)d_in[0],
        (const float*)d_in[3],  (const float*)d_in[4],
        (const float*)d_in[5],  (const float*)d_in[6],
        (const float*)d_in[7],  (const float*)d_in[8],
        (const float*)d_in[9],  (const float*)d_in[10],
        (const float*)d_in[11], (const float*)d_in[12],
        (const float*)d_in[13], (const float*)d_in[14],
        (const float*)d_in[15], (const float*)d_in[16],
        (float*)d_out, (char*)d_ws);
}
